// ReactionModality_90323162235725
// MI455X (gfx1250) — matrix operand order repaired and checked by reading, not compiled, not run on hardware
//
#include <hip/hip_runtime.h>
#include <math.h>

typedef __attribute__((ext_vector_type(16))) _Float16 v16h;
typedef __attribute__((ext_vector_type(8)))  float    v8f;

__device__ __forceinline__ float lrelu(float y) { return y > 0.f ? y : 0.01f * y; }

// ---------------- pack activations f32 [N,Di] -> f16 [N,Kp], zero-padded K ----
__global__ void pack_act_kernel(const float* __restrict__ x, _Float16* __restrict__ out,
                                int N, int Di, int Kp) {
  size_t i = (size_t)blockIdx.x * blockDim.x + threadIdx.x;
  size_t tot = (size_t)N * Kp;
  if (i >= tot) return;
  int k = (int)(i % Kp);
  size_t n = i / Kp;
  out[i] = (k < Di) ? (_Float16)x[n * Di + k] : (_Float16)0.f;
}

// ---- pack 4 weight mats [Do,Di] f32 -> WMMA B fragments, lane-major ----------
// Layout: wB[((m*CT + c)*KT + t)*32 + lane][16 halves]  (32B contiguous per lane)
// B 32x16 f16 fragment: lane<16 -> K=t*32+[0..15], lane>=16 -> K=t*32+[16..31];
// half h: j=h/2,pos=h&1 -> K += 2j+pos; column n = c*16 + lane%16.
__global__ void pack_wfrag_kernel(const float* __restrict__ w0, const float* __restrict__ w1,
                                  const float* __restrict__ w2, const float* __restrict__ w3,
                                  _Float16* __restrict__ wB, int Di, int Do, int Kp) {
  const int KT = Kp >> 5, CT = Do >> 4;
  const int tot = 4 * CT * KT * 32 * 16;
  int i = blockIdx.x * blockDim.x + threadIdx.x;
  if (i >= tot) return;
  const int h    = i & 15;
  const int lane = (i >> 4) & 31;
  const int rem  = i >> 9;          // ((m*CT + c)*KT + t)
  const int t  = rem % KT;
  const int mc = rem / KT;
  const int c  = mc % CT;
  const int m  = mc / CT;
  const int laneLo = lane & 15, hi = lane >> 4;
  const int k = (t << 5) + (hi << 4) + (h & ~1) + (h & 1);
  const int n = (c << 4) + laneLo;
  const float* Ws[4] = {w0, w1, w2, w3};
  wB[i] = (k < Di) ? (_Float16)Ws[m][n * Di + k] : (_Float16)0.f;
}

// ---------------- fused Q/K/V/S projection via WMMA ---------------------------
// Block: 256 thr = 8 waves; one 16-node row tile per block; wave -> (matrix, col-tile).
// A staged through LDS (2 x ds_load_b128 / K-tile); B pre-packed fragment-major
// (2 x global_load_b128 / K-tile); 1 x v_wmma_f32_16x16x32_f16 / K-tile.
template<int DO, int KP>
__global__ void gemm_qkvs_kernel(const _Float16* __restrict__ act,
                                 const _Float16* __restrict__ wB,
                                 const float* __restrict__ bq, const float* __restrict__ bk,
                                 const float* __restrict__ bv, const float* __restrict__ bs,
                                 float* __restrict__ q, float* __restrict__ k,
                                 float* __restrict__ v, float* __restrict__ s, int N) {
  __shared__ __align__(16) _Float16 As[16 * KP];
  const int r0 = blockIdx.x * 16;
  {
    const uint4* g = reinterpret_cast<const uint4*>(act) + (size_t)r0 * KP / 8;
    uint4* l = reinterpret_cast<uint4*>(As);
    const size_t gbase = (size_t)r0 * KP / 8;
    const size_t gmax  = (size_t)N * KP / 8;
    const int chunks = 16 * KP / 8;
    for (int i = threadIdx.x; i < chunks; i += blockDim.x) {
      uint4 z = make_uint4(0u, 0u, 0u, 0u);
      l[i] = (gbase + i < gmax) ? g[i] : z;
    }
  }
  __syncthreads();

  constexpr int CT = DO / 16;              // 16-col tiles per output matrix
  constexpr int KT = KP / 32;              // K tiles
  const int wave = threadIdx.x >> 5;
  const int lane = threadIdx.x & 31;
  if (wave >= 4 * CT) return;              // DO=16: waves 4..7 retire after sync
  const int m = wave / CT;
  const int c = wave % CT;
  const int laneLo = lane & 15, hi = lane >> 4;
  const int col = c * 16 + laneLo;

  // per-lane fragment base: ((m*CT + c)*KT + t)*32 + lane, 2 uint4 each
  const uint4* bbase = reinterpret_cast<const uint4*>(wB) +
                       ((size_t)(m * CT + c) * KT * 32 + lane) * 2;

  v8f acc = {};
  #pragma unroll
  for (int t = 0; t < KT; ++t) {
    union { v16h v; uint4 u[2]; } A, B;
    // A halves: e[0..15] = K [t*32+hi*16, +16)
    const _Float16* arow = &As[laneLo * KP + (t << 5) + (hi << 4)];
    A.u[0] = *reinterpret_cast<const uint4*>(arow);
    A.u[1] = *reinterpret_cast<const uint4*>(arow + 8);
    B.u[0] = bbase[t * 64];
    B.u[1] = bbase[t * 64 + 1];
    acc = __builtin_amdgcn_wmma_f32_16x16x32_f16(false, A.v, false, B.v,
                                                 (short)0, acc, false, false);
  }

  const float* biases[4] = {bq, bk, bv, bs};
  float* outs[4] = {q, k, v, s};
  const float bval = biases[m][col];
  float* O = outs[m];
  #pragma unroll
  for (int i = 0; i < 8; ++i) {           // C: VGPR i -> row (hi*8 + i), col = lane%16
    const int row = r0 + (hi << 3) + i;
    if (row < N) O[(size_t)row * DO + col] = acc[i] + bval;
  }
}

// ---------------- edge phase --------------------------------------------------
template<int D>
__global__ void edge_alpha_kernel(const float* __restrict__ q, const float* __restrict__ k,
                                  const int* __restrict__ src, const int* __restrict__ dst,
                                  float* __restrict__ alpha, int E, float scale) {
  int e = blockIdx.x * blockDim.x + threadIdx.x;
  if (e >= E) return;
  __builtin_prefetch(src + e + 8192, 0, 1);   // global_prefetch_b8: stream edge indices
  __builtin_prefetch(dst + e + 8192, 0, 1);
  const int sn = src[e], dn = dst[e];
  const float4* qd = reinterpret_cast<const float4*>(q + (size_t)dn * D);
  const float4* ks = reinterpret_cast<const float4*>(k + (size_t)sn * D);
  float acc = 0.f;
  #pragma unroll
  for (int i = 0; i < D / 4; ++i) {
    float4 a = qd[i], b = ks[i];
    acc += a.x * b.x + a.y * b.y + a.z * b.z + a.w * b.w;
  }
  alpha[e] = acc * scale;
}

__device__ __forceinline__ unsigned f2key(float f) {      // order-preserving map
  unsigned b = __float_as_uint(f);
  return (b & 0x80000000u) ? ~b : (b | 0x80000000u);
}
__device__ __forceinline__ float key2f(unsigned u) {
  return __uint_as_float((u & 0x80000000u) ? (u & 0x7FFFFFFFu) : ~u);
}

__global__ void edge_max_kernel(const float* __restrict__ alpha, const int* __restrict__ dst,
                                unsigned* __restrict__ mu, int E) {
  int e = blockIdx.x * blockDim.x + threadIdx.x;
  if (e >= E) return;
  atomicMax(&mu[dst[e]], f2key(alpha[e]));                // global_atomic_max_u32
}

__global__ void edge_expsum_kernel(float* __restrict__ alpha, const int* __restrict__ dst,
                                   const unsigned* __restrict__ mu, float* __restrict__ ssum,
                                   int E) {
  int e = blockIdx.x * blockDim.x + threadIdx.x;
  if (e >= E) return;
  const int dn = dst[e];
  const float ex = expf(alpha[e] - key2f(mu[dn]));
  alpha[e] = ex;
  atomicAdd(&ssum[dn], ex);                               // global_atomic_add_f32
}

// wave-per-edge weighted scatter-add: lane = feature (D=32) or half-wave (D=16)
template<int D>
__global__ void edge_agg_kernel(const float* __restrict__ alpha, const float* __restrict__ ssum,
                                const float* __restrict__ v, const int* __restrict__ src,
                                const int* __restrict__ dst, float* __restrict__ agg, int E) {
  constexpr int EPW = 32 / D;
  const int lane = threadIdx.x & 31;
  const long long wid = (long long)((blockIdx.x * blockDim.x + threadIdx.x) >> 5);
  const long long e = wid * EPW + (lane / D);
  const int f = lane % D;
  if (e >= E) return;
  const int sn = src[e], dn = dst[e];
  const float a = alpha[e] / (ssum[dn] + 1e-16f);
  atomicAdd(&agg[(size_t)dn * D + f], v[(size_t)sn * D + f] * a);
}

// ---------------- node epilogues ----------------------------------------------
// wave-per-node: out = lrelu(LayerNorm(agg + skip)); also emit f16 for next WMMA
__global__ void node_ln_kernel(const float* __restrict__ agg, const float* __restrict__ skip,
                               const float* __restrict__ g, const float* __restrict__ b,
                               _Float16* __restrict__ h16, int N) {
  const int lane = threadIdx.x & 31;
  const int n = (blockIdx.x * blockDim.x + threadIdx.x) >> 5;
  if (n >= N) return;
  const size_t idx = (size_t)n * 32 + lane;
  const float val = agg[idx] + skip[idx];
  float sm = val;
  for (int off = 16; off; off >>= 1) sm += __shfl_xor(sm, off, 32);
  const float mu_ = sm * (1.f / 32.f);
  const float d = val - mu_;
  float vs = d * d;
  for (int off = 16; off; off >>= 1) vs += __shfl_xor(vs, off, 32);
  float y = d * rsqrtf(vs * (1.f / 32.f) + 1e-5f) * g[lane] + b[lane];
  y = lrelu(y);
  h16[idx] = (_Float16)y;
}

// two nodes per wave (d=16): out = lrelu(L2normalize(agg + skip))
__global__ void node_l2_kernel(const float* __restrict__ agg, const float* __restrict__ skip,
                               float* __restrict__ out, int N) {
  const int lane = threadIdx.x & 31;
  const int wid = (blockIdx.x * blockDim.x + threadIdx.x) >> 5;
  const int n = wid * 2 + (lane >> 4);
  const int f = lane & 15;
  if (n >= N) return;
  const size_t idx = (size_t)n * 16 + f;
  const float val = agg[idx] + skip[idx];
  float sq = val * val;
  for (int off = 8; off; off >>= 1) sq += __shfl_xor(sq, off, 32);  // stays in 16-lane half
  const float nrm = sqrtf(sq);
  out[idx] = lrelu(val / fmaxf(nrm, 1e-12f));
}

// ---------------- host orchestration ------------------------------------------
extern "C" void kernel_launch(void* const* d_in, const int* in_sizes, int n_in,
                              void* d_out, int out_size, void* d_ws, size_t ws_size,
                              hipStream_t stream) {
  const int IN = 168;
  const int N = in_sizes[0] / IN;
  const int E = in_sizes[1] / 2;

  const float* x = (const float*)d_in[0];
  const int* src = (const int*)d_in[1];
  const int* dst = src + E;
  const float *W[3][4], *Bb[3][4];
  for (int l = 0; l < 3; ++l)
    for (int m = 0; m < 4; ++m) {
      W[l][m]  = (const float*)d_in[2 + l * 8 + m * 2];
      Bb[l][m] = (const float*)d_in[2 + l * 8 + m * 2 + 1];
    }
  const float* lnw[2] = {(const float*)d_in[26], (const float*)d_in[28]};
  const float* lnb[2] = {(const float*)d_in[27], (const float*)d_in[29]};

  char* p = (char*)d_ws;
  auto carve = [&](size_t bytes) { char* r = p; p += (bytes + 255) & ~(size_t)255; return (void*)r; };
  _Float16* act192 = (_Float16*)carve((size_t)N * 192 * 2);
  _Float16* act32  = (_Float16*)carve((size_t)N * 32 * 2);
  _Float16* wB     = (_Float16*)carve((size_t)4 * 192 * 32 * 2);
  float* q     = (float*)carve((size_t)N * 32 * 4);
  float* k     = (float*)carve((size_t)N * 32 * 4);
  float* v     = (float*)carve((size_t)N * 32 * 4);
  float* s     = (float*)carve((size_t)N * 32 * 4);
  float* alpha = (float*)carve((size_t)E * 4);
  unsigned* mu = (unsigned*)carve((size_t)N * 4);   // 0 <= every key: memset-friendly
  float* ssum  = (float*)carve((size_t)N * 4);
  float* agg   = (float*)carve((size_t)N * 32 * 4);
  (void)ws_size; (void)n_in; (void)out_size;

  const int TPB = 256;
  auto cdiv = [](long long a, long long b) { return (unsigned)((a + b - 1) / b); };
  const float isq32 = 0.17677669529663687f, isq16 = 0.25f;

  // ---- layer 1: 168 -> 32 ----
  pack_act_kernel<<<cdiv((long long)N * 192, TPB), TPB, 0, stream>>>(x, act192, N, IN, 192);
  pack_wfrag_kernel<<<cdiv(4 * 2 * 6 * 32 * 16, TPB), TPB, 0, stream>>>(
      W[0][0], W[0][1], W[0][2], W[0][3], wB, IN, 32, 192);
  gemm_qkvs_kernel<32, 192><<<cdiv(N, 16), TPB, 0, stream>>>(act192, wB,
      Bb[0][0], Bb[0][1], Bb[0][2], Bb[0][3], q, k, v, s, N);
  hipMemsetAsync(mu, 0, (size_t)N * 4, stream);
  hipMemsetAsync(ssum, 0, (size_t)N * 4, stream);
  hipMemsetAsync(agg, 0, (size_t)N * 32 * 4, stream);
  edge_alpha_kernel<32><<<cdiv(E, TPB), TPB, 0, stream>>>(q, k, src, dst, alpha, E, isq32);
  edge_max_kernel<<<cdiv(E, TPB), TPB, 0, stream>>>(alpha, dst, mu, E);
  edge_expsum_kernel<<<cdiv(E, TPB), TPB, 0, stream>>>(alpha, dst, mu, ssum, E);
  edge_agg_kernel<32><<<cdiv((long long)E * 32, TPB), TPB, 0, stream>>>(alpha, ssum, v, src, dst, agg, E);
  node_ln_kernel<<<cdiv((long long)N * 32, TPB), TPB, 0, stream>>>(agg, s, lnw[0], lnb[0], act32, N);

  // ---- layer 2: 32 -> 32 ----
  pack_wfrag_kernel<<<cdiv(4 * 2 * 1 * 32 * 16, TPB), TPB, 0, stream>>>(
      W[1][0], W[1][1], W[1][2], W[1][3], wB, 32, 32, 32);
  gemm_qkvs_kernel<32, 32><<<cdiv(N, 16), TPB, 0, stream>>>(act32, wB,
      Bb[1][0], Bb[1][1], Bb[1][2], Bb[1][3], q, k, v, s, N);
  hipMemsetAsync(mu, 0, (size_t)N * 4, stream);
  hipMemsetAsync(ssum, 0, (size_t)N * 4, stream);
  hipMemsetAsync(agg, 0, (size_t)N * 32 * 4, stream);
  edge_alpha_kernel<32><<<cdiv(E, TPB), TPB, 0, stream>>>(q, k, src, dst, alpha, E, isq32);
  edge_max_kernel<<<cdiv(E, TPB), TPB, 0, stream>>>(alpha, dst, mu, E);
  edge_expsum_kernel<<<cdiv(E, TPB), TPB, 0, stream>>>(alpha, dst, mu, ssum, E);
  edge_agg_kernel<32><<<cdiv((long long)E * 32, TPB), TPB, 0, stream>>>(alpha, ssum, v, src, dst, agg, E);
  node_ln_kernel<<<cdiv((long long)N * 32, TPB), TPB, 0, stream>>>(agg, s, lnw[1], lnb[1], act32, N);

  // ---- layer 3: 32 -> 16, L2-normalize epilogue into d_out ----
  pack_wfrag_kernel<<<cdiv(4 * 1 * 1 * 32 * 16, TPB), TPB, 0, stream>>>(
      W[2][0], W[2][1], W[2][2], W[2][3], wB, 32, 16, 32);
  gemm_qkvs_kernel<16, 32><<<cdiv(N, 16), TPB, 0, stream>>>(act32, wB,
      Bb[2][0], Bb[2][1], Bb[2][2], Bb[2][3], q, k, v, s, N);
  hipMemsetAsync(mu, 0, (size_t)N * 4, stream);
  hipMemsetAsync(ssum, 0, (size_t)N * 4, stream);
  hipMemsetAsync(agg, 0, (size_t)N * 16 * 4, stream);
  edge_alpha_kernel<16><<<cdiv(E, TPB), TPB, 0, stream>>>(q, k, src, dst, alpha, E, isq16);
  edge_max_kernel<<<cdiv(E, TPB), TPB, 0, stream>>>(alpha, dst, mu, E);
  edge_expsum_kernel<<<cdiv(E, TPB), TPB, 0, stream>>>(alpha, dst, mu, ssum, E);
  {
    long long waves = ((long long)E + 1) / 2;   // 2 edges per wave at D=16
    edge_agg_kernel<16><<<cdiv(waves * 32, TPB), TPB, 0, stream>>>(alpha, ssum, v, src, dst, agg, E);
  }
  {
    long long waves = ((long long)N + 1) / 2;   // 2 nodes per wave
    node_l2_kernel<<<cdiv(waves * 32, TPB), TPB, 0, stream>>>(agg, s, (float*)d_out, N);
  }
}
